// IndexerTopKFP8_15333033247348
// MI455X (gfx1250) — matrix operand order repaired and checked by reading, not compiled, not run on hardware
//
#include <hip/hip_runtime.h>

#define B_     32
#define H_     64
#define D_     128
#define PAGE_  64
#define PAGES_ 128
#define S_     8192
#define TOPK_  2048
#define STILE  128                   // s-positions per pipeline step (2 pages)
#define CHUNK  512                   // s-positions per workgroup
#define NSTEP  (CHUNK / STILE)       // 4 pipeline steps
#define PAGE_ELEMS (PAGE_ * D_)      // 8192 f32 per page (32 KB)
#define RAW_ELEMS  (STILE * D_)      // 16384 f32 per step buffer (64 KB)
#define SCALE_ 0.08838834764831845f

typedef __attribute__((ext_vector_type(16))) __bf16 v16bf;
typedef __attribute__((ext_vector_type(8)))  float  v8f;
typedef __attribute__((ext_vector_type(4)))  unsigned int v4u;
typedef __attribute__((ext_vector_type(8)))  int v8i;
typedef __attribute__((ext_vector_type(4)))  int v4i;

#if __has_builtin(__builtin_amdgcn_tensor_load_to_lds) && \
    __has_builtin(__builtin_amdgcn_s_wait_tensorcnt)
#define HAVE_TDM 1
#else
#define HAVE_TDM 0
#endif

union BFrag {
  v16bf v;
  unsigned int u[8];
};

// Native f32 -> bf16 (fptrunc, RNE); backend pairs these into
// v_cvt_pk_bf16_f32 where possible.
static __device__ __forceinline__ unsigned int pack2bf(float a, float b) {
  union { __bf16 h[2]; unsigned int u; } t;
  t.h[0] = (__bf16)a;
  t.h[1] = (__bf16)b;
  return t.u;
}

#if HAVE_TDM
// Issue one TDM transfer: one contiguous 8192-f32 page -> LDS (1-D tile).
// D# built per CDNA5 ISA ch.8: group0 = {count, lds_addr, global_addr, type=2},
// group1 = {data_size=4B, tensor_dim0=8192, tensor_dim1=1, tile_dim0=8192,
//           tile_dim1=1, tensor_dim0_stride=8192}. Groups 2/3 unused (zero).
static __device__ __forceinline__ void tdm_load_page(const float* gsrc,
                                                     unsigned lds_off) {
  unsigned long long ga = (unsigned long long)(uintptr_t)gsrc;
  v4u g0;
  g0.x = 1u;                                            // count=1 (valid)
  g0.y = lds_off;                                       // LDS byte address
  g0.z = (unsigned)(ga & 0xFFFFFFFFu);                  // global addr [31:0]
  g0.w = (unsigned)((ga >> 32) & 0x01FFFFFFu) | (2u << 30);  // [56:32]|type=2
  v8i g1;
  g1[0] = (2 << 16);                                    // data_size=4B, mask=0
  g1[1] = (int)((PAGE_ELEMS & 0xFFFFu) << 16);          // tensor_dim0[15:0]
  g1[2] = (int)(((PAGE_ELEMS >> 16) & 0xFFFFu) | (1u << 16)); // dim0 hi | tensor_dim1=1
  g1[3] = (int)((PAGE_ELEMS & 0xFFFFu) << 16);          // tile_dim0 = 8192
  g1[4] = 1;                                            // tile_dim1=1, tile_dim2=0
  g1[5] = PAGE_ELEMS;                                   // tensor_dim0_stride
  g1[6] = 0;
  g1[7] = 0;
  v4i z4 = {0, 0, 0, 0};
  v8i z8 = {0, 0, 0, 0, 0, 0, 0, 0};
  __builtin_amdgcn_tensor_load_to_lds(g0, g1, z4, z4, z8, 0);
}
#endif

// ---------------------------------------------------------------------------
// Kernel 1: scores[b, s] = sum_h w[b,h] * relu(q[b,h,:] . kv[b,s,:]) * SCALE
// One WG = one batch x 512-position chunk. TDM double-buffers raw fp32 pages
// into LDS; dequant + bf16 conversion happens at B-fragment build time; q is
// pre-swizzled in LDS into WMMA A-fragment order so each A-frag is one
// contiguous 32-byte v16bf load. 8 waves x 16-col strips x 4 M x 4 K WMMAs.
// ---------------------------------------------------------------------------
__global__ __launch_bounds__(256)
void indexer_scores_kernel(const float* __restrict__ q,
                           const float* __restrict__ kcur,
                           const float* __restrict__ w,
                           const float* __restrict__ k_cache,
                           const float* __restrict__ k_s_cache,
                           const int*   __restrict__ kv_seqlens,
                           const int*   __restrict__ block_offsets,
                           float*       __restrict__ scores_out) {
  __shared__ float          lds_raw[2][RAW_ELEMS];   // 2 x 64 KB raw fp32 pages
  __shared__ unsigned short lds_qf[H_ * D_];         // bf16 q, fragment order
  __shared__ float          lds_w[H_];

  const int b     = blockIdx.y;
  const int base0 = blockIdx.x * CHUNK;
  const int tid   = threadIdx.x;
  const int kvlen = kv_seqlens[b];

  // ---- stage q[b] as bf16 in A-fragment order ----
  // dst group index: ((h*4 + kk)*2 + hi8)*16 + pos0, 8 elements per group:
  //   sub=0: K kk*32+ 0..7  -> (hi8=0, pos0=0)   sub=1: + 8..15 -> (0, 8)
  //   sub=2: K kk*32+16..23 -> (hi8=1, pos0=0)   sub=3: +24..31 -> (1, 8)
  for (int g = tid; g < H_ * D_ / 8; g += 256) {
    const int h   = g >> 4;
    const int t8  = g & 15;
    const int kk  = t8 >> 2;
    const int sub = t8 & 3;
    const int hi8s = sub >> 1;
    const int pos0 = (sub & 1) * 8;
    const float* src = q + (size_t)b * H_ * D_ + h * D_ + kk * 32 + sub * 8;
    float4 v0 = *(const float4*)(src + 0);
    float4 v1 = *(const float4*)(src + 4);
    const int dst = ((h * 4 + kk) * 2 + hi8s) * 16 + pos0;
    *(uint4*)(&lds_qf[dst]) = make_uint4(pack2bf(v0.x, v0.y), pack2bf(v0.z, v0.w),
                                         pack2bf(v1.x, v1.y), pack2bf(v1.z, v1.w));
  }
  if (tid < H_) lds_w[tid] = w[b * H_ + tid];

#if HAVE_TDM
  // ---- pipeline prologue: DMA the two pages of step 0 ----
  if (tid < 32) {
    const int p0 = block_offsets[b * PAGES_ + (base0 >> 6)];
    const int p1 = block_offsets[b * PAGES_ + (base0 >> 6) + 1];
    tdm_load_page(k_cache + (size_t)p0 * PAGE_ELEMS,
                  (unsigned)(uintptr_t)&lds_raw[0][0]);
    tdm_load_page(k_cache + (size_t)p1 * PAGE_ELEMS,
                  (unsigned)(uintptr_t)&lds_raw[0][PAGE_ELEMS]);
  }
#endif

  const int lane = tid & 31;
  const int wave = tid >> 5;          // strip id: columns [wave*16, wave*16+16)
  const int col  = lane & 15;
  const int hi8  = lane >> 4;         // 0: lanes 0-15, 1: lanes 16-31

  for (int step = 0; step < NSTEP; ++step) {
    const int cur   = step & 1;
    const int cbase = base0 + step * STILE;

#if HAVE_TDM
    // issue step+1 into the other buffer (its readers finished last iter)
    if (tid < 32 && step + 1 < NSTEP) {
      const int nb = cbase + STILE;
      const int p0 = block_offsets[b * PAGES_ + (nb >> 6)];
      const int p1 = block_offsets[b * PAGES_ + (nb >> 6) + 1];
      tdm_load_page(k_cache + (size_t)p0 * PAGE_ELEMS,
                    (unsigned)(uintptr_t)&lds_raw[cur ^ 1][0]);
      tdm_load_page(k_cache + (size_t)p1 * PAGE_ELEMS,
                    (unsigned)(uintptr_t)&lds_raw[cur ^ 1][PAGE_ELEMS]);
    }
    if (step + 1 < NSTEP) __builtin_amdgcn_s_wait_tensorcnt(2);
    else                  __builtin_amdgcn_s_wait_tensorcnt(0);
    __syncthreads();                  // lds_raw[cur] ready for all waves
#else
    __syncthreads();
    for (int e = tid * 4; e < RAW_ELEMS; e += 256 * 4) {
      const int sl2 = e >> 7, d = e & (D_ - 1);
      const int sg2 = cbase + sl2;
      const int page2 = block_offsets[b * PAGES_ + (sg2 >> 6)];
      *(float4*)&lds_raw[cur][e] =
          *(const float4*)(k_cache + ((size_t)page2 * PAGE_ + (sg2 & 63)) * D_ + d);
    }
    __syncthreads();
#endif

    // ---- current-step key substitution (row kvlen-1, unscaled) ----
    const int curRow = kvlen - 1 - cbase;
    if (curRow >= 0 && curRow < STILE && tid < 32) {
      *(float4*)&lds_raw[cur][curRow * D_ + tid * 4] =
          *(const float4*)(kcur + (size_t)b * D_ + tid * 4);
    }
    __syncthreads();

    // ---- per-wave 16-column strip ----
    const int sl = wave * 16 + col;   // local column in step tile
    const int sg = cbase + sl;        // global s position
    const int page = block_offsets[b * PAGES_ + (sg >> 6)];
    const float sc = (sg == kvlen - 1)
                         ? 1.f
                         : k_s_cache[(size_t)page * PAGE_ + (sg & 63)];

    v8f acc[4] = {v8f{}, v8f{}, v8f{}, v8f{}};

    #pragma unroll
    for (int kk = 0; kk < D_ / 32; ++kk) {
      const int k0 = kk * 32;
      // B fragment: dequantize + convert 16 contiguous f32 -> v16bf inline.
      const float* rp = &lds_raw[cur][sl * D_ + k0 + hi8 * 16];
      float4 r0 = *(const float4*)(rp + 0);
      float4 r1 = *(const float4*)(rp + 4);
      float4 r2 = *(const float4*)(rp + 8);
      float4 r3 = *(const float4*)(rp + 12);
      BFrag bf;
      bf.u[0] = pack2bf(r0.x * sc, r0.y * sc);
      bf.u[1] = pack2bf(r0.z * sc, r0.w * sc);
      bf.u[2] = pack2bf(r1.x * sc, r1.y * sc);
      bf.u[3] = pack2bf(r1.z * sc, r1.w * sc);
      bf.u[4] = pack2bf(r2.x * sc, r2.y * sc);
      bf.u[5] = pack2bf(r2.z * sc, r2.w * sc);
      bf.u[6] = pack2bf(r3.x * sc, r3.y * sc);
      bf.u[7] = pack2bf(r3.z * sc, r3.w * sc);
      #pragma unroll
      for (int m = 0; m < 4; ++m) {
        // A fragment: one contiguous 32-byte read in fragment order.
        const v16bf af = *(const v16bf*)(
            &lds_qf[(((m * 16 + col) * 4 + kk) * 2 + hi8) * 16]);
        acc[m] = __builtin_amdgcn_wmma_f32_16x16x32_bf16(
            false, af, false, bf.v, (short)0, acc[m], false, false);
      }
    }

    // ---- relu + per-head weights + fold lane halves ----
    float partial = 0.f;
    #pragma unroll
    for (int m = 0; m < 4; ++m) {
      #pragma unroll
      for (int r = 0; r < 8; ++r) {
        partial += lds_w[m * 16 + r + hi8 * 8] * fmaxf(acc[m][r], 0.f);
      }
    }
    partial += __shfl_xor(partial, 16, 32);

    if (lane < 16) {
      const float v = (sg < kvlen) ? partial * SCALE_ : -__builtin_inff();
      scores_out[(size_t)b * S_ + sg] = v;
    }
    __syncthreads();                  // all reads of lds_raw[cur] done
  }
}

// ---------------------------------------------------------------------------
// Kernel 2: per-batch full bitonic sort (descending) of 8192 (score, idx)
// pairs in LDS; emit top 2048 indices, fill -1 where score == -inf.
// ---------------------------------------------------------------------------
__global__ __launch_bounds__(1024)
void indexer_topk_kernel(const float* __restrict__ scores,
                         int*         __restrict__ idx_out) {
  __shared__ float sv[S_];
  __shared__ int   si[S_];

  const int b   = blockIdx.x;
  const int tid = threadIdx.x;

  for (int i = tid; i < S_; i += 1024) {
    sv[i] = scores[(size_t)b * S_ + i];
    si[i] = i;
  }
  __syncthreads();

  for (int kk = 2; kk <= S_; kk <<= 1) {
    for (int j = kk >> 1; j > 0; j >>= 1) {
      for (int i = tid; i < S_; i += 1024) {
        const int ixj = i ^ j;
        if (ixj > i) {
          const bool desc = ((i & kk) == 0);     // final pass: all descending
          const float a = sv[i], c = sv[ixj];
          if (desc ? (a < c) : (a > c)) {
            sv[i] = c; sv[ixj] = a;
            const int t = si[i]; si[i] = si[ixj]; si[ixj] = t;
          }
        }
      }
      __syncthreads();
    }
  }

  for (int t = tid; t < TOPK_; t += 1024) {
    const float v = sv[t];
    idx_out[(size_t)b * TOPK_ + t] = (v == -__builtin_inff()) ? -1 : si[t];
  }
}

// ---------------------------------------------------------------------------
extern "C" void kernel_launch(void* const* d_in, const int* in_sizes, int n_in,
                              void* d_out, int out_size, void* d_ws, size_t ws_size,
                              hipStream_t stream) {
  (void)in_sizes; (void)n_in; (void)out_size; (void)d_ws; (void)ws_size;
  const float* q             = (const float*)d_in[0];
  const float* k             = (const float*)d_in[1];
  const float* weights       = (const float*)d_in[2];
  const float* k_cache       = (const float*)d_in[3];
  const float* k_s_cache     = (const float*)d_in[4];
  const int*   kv_seqlens    = (const int*)d_in[5];
  const int*   block_offsets = (const int*)d_in[6];

  int*   idx_out    = (int*)d_out;                    // B*TOPK int32 (first)
  float* scores_out = (float*)d_out + B_ * TOPK_;     // B*S float32 (second)

  dim3 g1(S_ / CHUNK, B_);
  indexer_scores_kernel<<<g1, 256, 0, stream>>>(
      q, k, weights, k_cache, k_s_cache, kv_seqlens, block_offsets, scores_out);

  indexer_topk_kernel<<<B_, 1024, 0, stream>>>(scores_out, idx_out);
}
